// LSTM_75307956568763
// MI455X (gfx1250) — matrix operand order repaired and checked by reading, not compiled, not run on hardware
//
#include <hip/hip_runtime.h>
#include <hip/hip_bf16.h>
#include <stdint.h>

// ---------------------------------------------------------------------------
// Two-layer LSTM (seq_len=1, h0=c0=0) + classifier, fused for MI455X (gfx1250)
//   GEMM1: x[8192,1024] @ W_ih1^T (i,g,o blocks only) -> h1 (bf16)
//   GEMM2: h1 @ W_ih2^T (i,g,o) -> h2 (bf16)
//   GEMM3: h2 @ W_cls^T + b_cls -> out (f32)
// v_wmma_f32_16x16x32_bf16 (wave32) GEMM cores, fp32 accumulate.
// Global->LDS staging via GLOBAL_LOAD_ASYNC_TO_LDS_B128 (ASYNCcnt),
// double-buffered with the steady-state loop fully branch-free.
// ---------------------------------------------------------------------------

typedef __bf16 bf16_t;
typedef bf16_t bf16x16 __attribute__((ext_vector_type(16)));
typedef float  f32x8   __attribute__((ext_vector_type(8)));
typedef int    v4i_vs  __attribute__((__vector_size__(16)));

union FragU { uint4 q[2]; bf16x16 v; };

static constexpr int BATCH = 8192;
static constexpr int DIN   = 1024;
static constexpr int HID   = 2048;
static constexpr int NCLS  = 1000;

#if __has_builtin(__builtin_amdgcn_global_load_async_to_lds_b128)
#define USE_ASYNC_LDS 1
#else
#define USE_ASYNC_LDS 0
#endif

#if USE_ASYNC_LDS
#if __has_builtin(__builtin_amdgcn_s_wait_asynccnt)
#define WAIT_ASYNC(n) __builtin_amdgcn_s_wait_asynccnt(n)
#else
#define WAIT_ASYNC(n) asm volatile("s_wait_asynccnt %0" ::"i"(n) : "memory")
#endif
#else
#define WAIT_ASYNC(n)
#endif

// copy 16 bytes global -> LDS (async DMA if available, else through VGPRs)
static __device__ __forceinline__ void cp16(const bf16_t* g, bf16_t* l) {
#if USE_ASYNC_LDS
    __builtin_amdgcn_global_load_async_to_lds_b128(
        (__attribute__((address_space(1))) v4i_vs*)g,
        (__attribute__((address_space(3))) v4i_vs*)l, 0, 0);
#else
    *reinterpret_cast<uint4*>(l) = *reinterpret_cast<const uint4*>(g);
#endif
}

static __device__ __forceinline__ unsigned short bf16_bits(float f) {
    return __builtin_bit_cast(unsigned short, (bf16_t)f);
}
static __device__ __forceinline__ float sigm(float x) {
    return 1.0f / (1.0f + __expf(-x));
}
static __device__ __forceinline__ float tanh_f(float x) {
    return 2.0f * sigm(2.0f * x) - 1.0f;
}

// ---------------- conversion / packing kernels ----------------

__global__ void cvt_f32_bf16_kernel(const float* __restrict__ src,
                                    bf16_t* __restrict__ dst, int n4) {
    int i = blockIdx.x * blockDim.x + threadIdx.x;
    if (i >= n4) return;
    float4 f = reinterpret_cast<const float4*>(src)[i];
    ushort4 u = make_ushort4(bf16_bits(f.x), bf16_bits(f.y),
                             bf16_bits(f.z), bf16_bits(f.w));
    reinterpret_cast<ushort4*>(dst)[i] = u;
}

// Pack rows {0..2047, 4096..6143, 6144..8191} (i,g,o) of W[8192,K] into
// bf16 Wp[3*2048, K]. kshift = log2(K).
__global__ void pack_gates_kernel(const float* __restrict__ w,
                                  bf16_t* __restrict__ dst,
                                  int kshift, int n4) {
    int i = blockIdx.x * blockDim.x + threadIdx.x;
    if (i >= n4) return;
    int K = 1 << kshift;
    int flat = i << 2;
    int row  = flat >> kshift;     // 0..6143
    int k    = flat & (K - 1);
    int g    = row >> 11;          // 0,1,2
    int r    = row & 2047;
    int srow = (g == 0 ? 0 : (g == 1 ? 4096 : 6144)) + r;
    float4 f = *reinterpret_cast<const float4*>(w + (size_t)srow * K + k);
    ushort4 u = make_ushort4(bf16_bits(f.x), bf16_bits(f.y),
                             bf16_bits(f.z), bf16_bits(f.w));
    *reinterpret_cast<ushort4*>(dst + (size_t)row * K + k) = u;
}

// W_cls[1000,2048] -> bf16 [1024,2048], zero-padded rows 1000..1023.
__global__ void pack_cls_kernel(const float* __restrict__ w,
                                bf16_t* __restrict__ dst, int n4) {
    int i = blockIdx.x * blockDim.x + threadIdx.x;
    if (i >= n4) return;
    int row = i >> 9;
    int k4  = i & 511;
    ushort4 u = make_ushort4(0, 0, 0, 0);
    if (row < NCLS) {
        float4 f = *reinterpret_cast<const float4*>(w + (size_t)row * HID + k4 * 4);
        u = make_ushort4(bf16_bits(f.x), bf16_bits(f.y),
                         bf16_bits(f.z), bf16_bits(f.w));
    }
    *reinterpret_cast<ushort4*>(dst + (size_t)row * HID + k4 * 4) = u;
}

// ---------------- fused LSTM-layer GEMM ----------------
// A [8192, K] bf16 row-major; Wp [3*2048, K] bf16 (i,g,o packed, row = out col);
// b_ih/b_hh fp32 [8192] original gate layout; Hout [8192, 2048] bf16.
// 256 threads = 8 waves (4 M x 2 N); WG tile 128x64; wave tile 32x32;
// K-tile 32, double-buffered async global->LDS staging, peeled last iter.
__global__ __launch_bounds__(256)
void lstm_layer_gemm_kernel(const bf16_t* __restrict__ A,
                            const bf16_t* __restrict__ Wp,
                            const float* __restrict__ b_ih,
                            const float* __restrict__ b_hh,
                            bf16_t* __restrict__ Hout, int K) {
    __shared__ alignas(16) bf16_t As[2][128][40];
    __shared__ alignas(16) bf16_t Bs[2][3][64][40];

    const int tid  = threadIdx.x;
    const int lane = tid & 31;
    const int w    = tid >> 5;
    const int wr   = w & 3;        // wave M index (0..3)
    const int wcn  = w >> 2;       // wave N index (0..1)
    const int nlo  = lane & 15;
    const int half = lane >> 4;

    const int m0 = blockIdx.y * 128;   // batch row base
    const int n0 = blockIdx.x * 64;    // h-column base (0..2047)

    f32x8 acc[3][2][2];
#pragma unroll
    for (int g = 0; g < 3; ++g)
#pragma unroll
        for (int mi = 0; mi < 2; ++mi)
#pragma unroll
            for (int ni = 0; ni < 2; ++ni)
#pragma unroll
                for (int v = 0; v < 8; ++v) acc[g][mi][ni][v] = 0.0f;

    // stage K-tile k0 into buffer `buf` (5 x 16B async copies per thread)
    auto stage = [&](int buf, int k0) {
#pragma unroll
        for (int i = 0; i < 2; ++i) {            // A: 128 rows x 32 halves
            int idx = i * 256 + tid;
            int r = idx >> 2, c = (idx & 3) * 8;
            cp16(A + (size_t)(m0 + r) * K + k0 + c, &As[buf][r][c]);
        }
#pragma unroll
        for (int i = 0; i < 3; ++i) {            // B: 3 gates x 64 rows x 32 halves
            int idx = i * 256 + tid;
            int g = idx >> 8, rem = idx & 255;
            int r = rem >> 2, c = (rem & 3) * 8;
            cp16(Wp + (size_t)(g * 2048 + n0 + r) * K + k0 + c, &Bs[buf][g][r][c]);
        }
    };

    // load all fragments first, then run the 12 WMMAs back-to-back
    auto compute = [&](int cur) {
        FragU fa[2];
        FragU fb[3][2];
#pragma unroll
        for (int mi = 0; mi < 2; ++mi) {
            int ar = wr * 32 + mi * 16 + nlo;
            fa[mi].q[0] = *reinterpret_cast<const uint4*>(&As[cur][ar][8 * half]);
            fa[mi].q[1] = *reinterpret_cast<const uint4*>(&As[cur][ar][16 + 8 * half]);
        }
#pragma unroll
        for (int g = 0; g < 3; ++g)
#pragma unroll
            for (int ni = 0; ni < 2; ++ni) {
                int br = wcn * 32 + ni * 16 + nlo;
                fb[g][ni].q[0] = *reinterpret_cast<const uint4*>(&Bs[cur][g][br][8 * half]);
                fb[g][ni].q[1] = *reinterpret_cast<const uint4*>(&Bs[cur][g][br][16 + 8 * half]);
            }
#pragma unroll
        for (int g = 0; g < 3; ++g)
#pragma unroll
            for (int ni = 0; ni < 2; ++ni)
#pragma unroll
                for (int mi = 0; mi < 2; ++mi)
                    acc[g][mi][ni] = __builtin_amdgcn_wmma_f32_16x16x32_bf16(
                        false, fa[mi].v, false, fb[g][ni].v,
                        (short)0, acc[g][mi][ni], false, false);
    };

    const int T = K >> 5;          // number of 32-wide K tiles (>= 32)
    stage(0, 0);
    for (int t = 0; t < T - 1; ++t) {
        const int cur = t & 1;
        stage(cur ^ 1, (t + 1) << 5);
        WAIT_ASYNC(5);             // tile t's 5 copies done; next tile in flight
        __syncthreads();
        compute(cur);
        __syncthreads();           // protect WAR on the buffer staged next iter
    }
    WAIT_ASYNC(0);
    __syncthreads();
    compute((T - 1) & 1);

    // fused epilogue: h = tanh( sigm(o) * tanh( sigm(i) * tanh(g) ) )
    const int mw = m0 + wr * 32;
    const int nw = n0 + wcn * 32;
#pragma unroll
    for (int ni = 0; ni < 2; ++ni) {
        int col = nw + ni * 16 + nlo;
        float bi = b_ih[col]        + b_hh[col];
        float bg = b_ih[col + 4096] + b_hh[col + 4096];
        float bo = b_ih[col + 6144] + b_hh[col + 6144];
#pragma unroll
        for (int mi = 0; mi < 2; ++mi) {
#pragma unroll
            for (int v = 0; v < 8; ++v) {
                int row = mw + mi * 16 + half * 8 + v;
                float gi = sigm(acc[0][mi][ni][v] + bi);
                float gg = tanh_f(acc[1][mi][ni][v] + bg);
                float go = sigm(acc[2][mi][ni][v] + bo);
                float c  = gi * gg;
                float h  = go * tanh_f(c);
                Hout[(size_t)row * HID + col] = (bf16_t)tanh_f(h);
            }
        }
    }
}

// ---------------- classifier GEMM ----------------
// out[8192,1000] f32 = h2[8192,2048] @ Wc^T + b_cls; Wc packed bf16 [1024,2048].
__global__ __launch_bounds__(256)
void cls_gemm_kernel(const bf16_t* __restrict__ A,
                     const bf16_t* __restrict__ Wc,
                     const float* __restrict__ bias,
                     float* __restrict__ out) {
    __shared__ alignas(16) bf16_t As[2][128][40];
    __shared__ alignas(16) bf16_t Bs[2][64][40];

    const int tid  = threadIdx.x;
    const int lane = tid & 31;
    const int w    = tid >> 5;
    const int wr   = w & 3;
    const int wcn  = w >> 2;
    const int nlo  = lane & 15;
    const int half = lane >> 4;
    const int K    = HID;

    const int m0 = blockIdx.y * 128;
    const int n0 = blockIdx.x * 64;

    f32x8 acc[2][2];
#pragma unroll
    for (int mi = 0; mi < 2; ++mi)
#pragma unroll
        for (int ni = 0; ni < 2; ++ni)
#pragma unroll
            for (int v = 0; v < 8; ++v) acc[mi][ni][v] = 0.0f;

    auto stage = [&](int buf, int k0) {
#pragma unroll
        for (int i = 0; i < 2; ++i) {
            int idx = i * 256 + tid;
            int r = idx >> 2, c = (idx & 3) * 8;
            cp16(A + (size_t)(m0 + r) * K + k0 + c, &As[buf][r][c]);
        }
        {
            int r = tid >> 2, c = (tid & 3) * 8;
            cp16(Wc + (size_t)(n0 + r) * K + k0 + c, &Bs[buf][r][c]);
        }
    };

    auto compute = [&](int cur) {
        FragU fa[2];
        FragU fb[2];
#pragma unroll
        for (int mi = 0; mi < 2; ++mi) {
            int ar = wr * 32 + mi * 16 + nlo;
            fa[mi].q[0] = *reinterpret_cast<const uint4*>(&As[cur][ar][8 * half]);
            fa[mi].q[1] = *reinterpret_cast<const uint4*>(&As[cur][ar][16 + 8 * half]);
        }
#pragma unroll
        for (int ni = 0; ni < 2; ++ni) {
            int br = wcn * 32 + ni * 16 + nlo;
            fb[ni].q[0] = *reinterpret_cast<const uint4*>(&Bs[cur][br][8 * half]);
            fb[ni].q[1] = *reinterpret_cast<const uint4*>(&Bs[cur][br][16 + 8 * half]);
        }
#pragma unroll
        for (int ni = 0; ni < 2; ++ni)
#pragma unroll
            for (int mi = 0; mi < 2; ++mi)
                acc[mi][ni] = __builtin_amdgcn_wmma_f32_16x16x32_bf16(
                    false, fa[mi].v, false, fb[ni].v,
                    (short)0, acc[mi][ni], false, false);
    };

    const int T = K >> 5;
    stage(0, 0);
    for (int t = 0; t < T - 1; ++t) {
        const int cur = t & 1;
        stage(cur ^ 1, (t + 1) << 5);
        WAIT_ASYNC(3);
        __syncthreads();
        compute(cur);
        __syncthreads();
    }
    WAIT_ASYNC(0);
    __syncthreads();
    compute((T - 1) & 1);

    const int mw = m0 + wr * 32;
    const int nw = n0 + wcn * 32;
#pragma unroll
    for (int ni = 0; ni < 2; ++ni) {
        int col = nw + ni * 16 + nlo;
        if (col < NCLS) {
            float bc = bias[col];
#pragma unroll
            for (int mi = 0; mi < 2; ++mi) {
#pragma unroll
                for (int v = 0; v < 8; ++v) {
                    int row = mw + mi * 16 + half * 8 + v;
                    out[(size_t)row * NCLS + col] = acc[mi][ni][v] + bc;
                }
            }
        }
    }
}

// ---------------- host side ----------------

extern "C" void kernel_launch(void* const* d_in, const int* in_sizes, int n_in,
                              void* d_out, int out_size, void* d_ws, size_t ws_size,
                              hipStream_t stream) {
    (void)in_sizes; (void)n_in; (void)out_size; (void)ws_size;

    const float* x     = (const float*)d_in[0];
    const float* W_ih1 = (const float*)d_in[1];
    const float* b_ih1 = (const float*)d_in[3];
    const float* b_hh1 = (const float*)d_in[4];
    const float* W_ih2 = (const float*)d_in[5];
    const float* b_ih2 = (const float*)d_in[7];
    const float* b_hh2 = (const float*)d_in[8];
    const float* W_cls = (const float*)d_in[9];
    const float* b_cls = (const float*)d_in[10];
    float* out = (float*)d_out;

    char* ws = (char*)d_ws;
    auto alignup = [](size_t v) { return (v + 255) & ~(size_t)255; };
    size_t off = 0;
    bf16_t* xb = (bf16_t*)(ws + off); off = alignup(off + (size_t)BATCH * DIN * 2);
    bf16_t* w1 = (bf16_t*)(ws + off); off = alignup(off + (size_t)6144 * DIN * 2);
    bf16_t* w2 = (bf16_t*)(ws + off); off = alignup(off + (size_t)6144 * HID * 2);
    bf16_t* wc = (bf16_t*)(ws + off); off = alignup(off + (size_t)1024 * HID * 2);
    bf16_t* h1 = (bf16_t*)(ws + off); off = alignup(off + (size_t)BATCH * HID * 2);
    bf16_t* h2 = (bf16_t*)(ws + off); off = alignup(off + (size_t)BATCH * HID * 2);

    {   // x -> bf16
        int n4 = BATCH * DIN / 4;
        cvt_f32_bf16_kernel<<<n4 / 256, 256, 0, stream>>>(x, xb, n4);
    }
    {   // pack W_ih1 (i,g,o rows)
        int n4 = 6144 * DIN / 4;
        pack_gates_kernel<<<n4 / 256, 256, 0, stream>>>(W_ih1, w1, 10, n4);
    }
    {   // pack W_ih2
        int n4 = 6144 * HID / 4;
        pack_gates_kernel<<<n4 / 256, 256, 0, stream>>>(W_ih2, w2, 11, n4);
    }
    {   // pack W_cls
        int n4 = 1024 * HID / 4;
        pack_cls_kernel<<<n4 / 256, 256, 0, stream>>>(W_cls, wc, n4);
    }
    {   // layer 1
        dim3 grid(HID / 64, BATCH / 128);
        lstm_layer_gemm_kernel<<<grid, 256, 0, stream>>>(xb, w1, b_ih1, b_hh1, h1, DIN);
    }
    {   // layer 2
        dim3 grid(HID / 64, BATCH / 128);
        lstm_layer_gemm_kernel<<<grid, 256, 0, stream>>>(h1, w2, b_ih2, b_hh2, h2, HID);
    }
    {   // classifier
        dim3 grid(1024 / 64, BATCH / 128);
        cls_gemm_kernel<<<grid, 256, 0, stream>>>(h2, wc, b_cls, out);
    }
}
